// Encoder_62792421867711
// MI455X (gfx1250) — matrix operand order repaired and checked by reading, not compiled, not run on hardware
//
#include <hip/hip_runtime.h>
#include <stdint.h>

// ---------------------------------------------------------------------------
// 3-layer BiLSTM encoder for MI455X (gfx1250, wave32, WMMA).
//  Phase A (per layer): token-parallel GEMM  Xg = A_f16 @ Wi_f16 + b   (WMMA),
//                       32x64 macro-tile per wave (8 WMMA tiles, 1.5 loads per
//                       WMMA), output time-major [T][B][1024].
//  Phase B (per layer): persistent-workgroup recurrent scan, one WG per
//                       direction; h in LDS (f16), c in registers, h@Wh via
//                       v_wmma_f32_16x16x32_f16, gates fused in-register,
//                       Xg loads hoisted ahead of the WMMA chain + software
//                       prefetch (global_prefetch_b8) of the next slice.
// ---------------------------------------------------------------------------

typedef __attribute__((ext_vector_type(16))) _Float16 v16h;
typedef __attribute__((ext_vector_type(8)))  float    v8f;
typedef __attribute__((ext_vector_type(4)))  unsigned int u32x4;

#define T_SEQ  512
#define B_SZ   32
#define UNITS  256
#define GATES  1024   // 4*UNITS
#define EMB_D  300
#define NTOK   (B_SZ * T_SEQ)   // 16384

union V16H { v16h v; u32x4 q[2]; };

__device__ __forceinline__ float sigmoidf_(float x) {
  return 1.0f / (1.0f + __expf(-x));
}

// ---------------------------------------------------------------------------
// Embedding lookup + cast to f16, zero-padded K: A[NTOK x 320] (token-major)
// ---------------------------------------------------------------------------
__global__ void embed_cast_kernel(const int* __restrict__ x,
                                  const float* __restrict__ emb,
                                  _Float16* __restrict__ A, int n) {
  int i = blockIdx.x * blockDim.x + threadIdx.x;
  if (i >= n) return;
  int tok = i / 320;
  int e   = i - tok * 320;
  float v = (e < EMB_D) ? emb[(long)x[tok] * EMB_D + e] : 0.0f;
  A[i] = (_Float16)v;
}

// ---------------------------------------------------------------------------
// Weight transpose + cast:  W[K x N] f32  ->  WT[N x Kpad] f16 (zero padded).
// Row n of WT holds column n of W contiguously -> B-fragment loads become
// two 16B chunks per lane, +32B apart.
// ---------------------------------------------------------------------------
__global__ void wcast_kernel(const float* __restrict__ W,
                             _Float16* __restrict__ WT,
                             int K, int N, int Kpad) {
  int i = blockIdx.x * blockDim.x + threadIdx.x;
  if (i >= N * Kpad) return;
  int nn = i / Kpad;
  int k  = i - nn * Kpad;
  WT[i] = (k < K) ? (_Float16)W[(long)k * N + nn] : (_Float16)0.0f;
}

// ---------------------------------------------------------------------------
// Phase A GEMM: Xg = A(f16, [NTOK x lda]) x WT^T + bias, both directions via
// blockIdx.y. 8 waves/block; each wave owns a 32x64 macro-tile:
// 2 M-tiles x 4 N-tiles = 8 WMMA accumulators. Per K-step: 4 A-chunk loads +
// 8 B-chunk loads feed 8 WMMAs. Output time-major: Xg[(t*B + b)*GATES + col].
// ---------------------------------------------------------------------------
__global__ __launch_bounds__(256) void xproj_gemm_kernel(
    const _Float16* __restrict__ A, int lda,
    const _Float16* __restrict__ WTf, const _Float16* __restrict__ WTb,
    const float* __restrict__ bf, const float* __restrict__ bb,
    float* __restrict__ Xgf, float* __restrict__ Xgb, int ktiles)
{
  const _Float16* WT = blockIdx.y ? WTb : WTf;
  const float* bias  = blockIdx.y ? bb  : bf;
  float* Xg          = blockIdx.y ? Xgb : Xgf;

  int wave = threadIdx.x >> 5;
  int lane = threadIdx.x & 31;
  int job  = blockIdx.x * 8 + wave;   // 0..8191
  int mb   = job >> 4;                // 0..511  -> rows mb*32
  int nb   = job & 15;                // 0..15   -> cols nb*64
  int l16  = lane & 15;
  int hlf  = lane >> 4;               // 0 or 1

  const _Float16* arow0 = A + (long)(mb * 32 + l16) * lda;   // M-tile 0
  const _Float16* arow1 = arow0 + (long)16 * lda;            // M-tile 1
  const _Float16* brow[4];
#pragma unroll
  for (int j = 0; j < 4; ++j)
    brow[j] = WT + (long)((nb * 4 + j) * 16 + l16) * lda + hlf * 8;

  v8f acc[2][4];
#pragma unroll
  for (int mi = 0; mi < 2; ++mi)
#pragma unroll
    for (int j = 0; j < 4; ++j) { v8f z = {}; acc[mi][j] = z; }

  for (int kt = 0; kt < ktiles; ++kt) {
    V16H a0, a1;
    const _Float16* ap0 = arow0 + kt * 32 + hlf * 8;   // two 16B chunks, +32B apart
    a0.q[0] = *(const u32x4*)(ap0);
    a0.q[1] = *(const u32x4*)(ap0 + 16);
    const _Float16* ap1 = arow1 + kt * 32 + hlf * 8;
    a1.q[0] = *(const u32x4*)(ap1);
    a1.q[1] = *(const u32x4*)(ap1 + 16);
#pragma unroll
    for (int j = 0; j < 4; ++j) {
      V16H b;
      const _Float16* bp = brow[j] + kt * 32;          // two 16B chunks, +32B apart
      b.q[0] = *(const u32x4*)(bp);
      b.q[1] = *(const u32x4*)(bp + 16);
      acc[0][j] = __builtin_amdgcn_wmma_f32_16x16x32_f16(false, a0.v, false, b.v,
                                                         (short)0, acc[0][j],
                                                         false, false);
      acc[1][j] = __builtin_amdgcn_wmma_f32_16x16x32_f16(false, a1.v, false, b.v,
                                                         (short)0, acc[1][j],
                                                         false, false);
    }
  }

#pragma unroll
  for (int mi = 0; mi < 2; ++mi) {
#pragma unroll
    for (int j = 0; j < 4; ++j) {
      int col  = (nb * 4 + j) * 16 + l16;
      float bv = bias[col];
#pragma unroll
      for (int r = 0; r < 8; ++r) {
        int m = mb * 32 + mi * 16 + r + hlf * 8;   // D layout: row = r+8*(lane/16)
        int b = m >> 9;                            // token = b*T + t
        int t = m & (T_SEQ - 1);
        Xg[((long)t * B_SZ + b) * GATES + col] = acc[mi][j][r] + bv;
      }
    }
  }
}

// ---------------------------------------------------------------------------
// Phase B: recurrent scan. grid.x = 2 (dir 0 = "f": scans reversed input;
// dir 1 = "b": scans forward). 1024 threads = 32 waves. Wave w owns M-tile
// mt=w/16 and unit-tile ut=w%16 across all four gates (N-tiles ut+16*g), so
// i,f,g,o for each (batch,unit) live in that wave's 4 accumulators -> gate
// math is in-register. h: LDS f16 [32x256]; c: 8 floats per lane.
// Xg/y are time-major.
// ---------------------------------------------------------------------------
__global__ __launch_bounds__(1024) void lstm_scan_kernel(
    const float* __restrict__ Xgf, const float* __restrict__ Xgb,
    const _Float16* __restrict__ WhTf, const _Float16* __restrict__ WhTb,
    const float* __restrict__ stateIn,   // [hf,cf,hb,cb] x 8192 each, or null
    float* __restrict__ yf, float* __restrict__ yb,
    float* __restrict__ stateOut)        // same layout
{
  __shared__ _Float16 hbuf[B_SZ * UNITS];   // 16 KB

  int dir = blockIdx.x;
  const float*    Xg  = dir ? Xgb  : Xgf;
  const _Float16* WhT = dir ? WhTb : WhTf;
  float*          y   = dir ? yb   : yf;
  const float* h0 = stateIn ? stateIn + (long)dir * 2 * 8192        : nullptr;
  const float* c0 = stateIn ? stateIn + (long)dir * 2 * 8192 + 8192 : nullptr;
  const int rev = (dir == 0);

  int tid  = threadIdx.x;
  int wave = tid >> 5;
  int lane = tid & 31;
  int mt   = wave >> 4;        // 0..1
  int ut   = wave & 15;        // 0..15
  int l16  = lane & 15;
  int hlf  = lane >> 4;
  int u    = ut * 16 + l16;    // this lane's unit column

  // init h
  for (int i = tid; i < B_SZ * UNITS; i += 1024)
    hbuf[i] = h0 ? (_Float16)h0[i] : (_Float16)0.0f;

  // init c (registers)
  float creg[8];
#pragma unroll
  for (int r = 0; r < 8; ++r) {
    int b = mt * 16 + r + hlf * 8;
    creg[r] = c0 ? c0[b * UNITS + u] : 0.0f;
  }
  __syncthreads();

  // B-fragment base pointers for the 4 gate tiles of this wave
  const _Float16* brow[4];
#pragma unroll
  for (int g = 0; g < 4; ++g)
    brow[g] = WhT + (long)(g * UNITS + u) * UNITS + hlf * 8;

  const int arowoff = (mt * 16 + l16) * UNITS + hlf * 8;
  const int brow0   = mt * 16 + hlf * 8;     // first batch row of this lane

  for (int s = 0; s < T_SEQ; ++s) {
    int t_in  = rev ? (T_SEQ - 1 - s) : s;
    int t_out = T_SEQ - 1 - t_in;

    // Prefetch next step's contiguous 128KB Xg slice (one 128B line/thread).
    if (s + 1 < T_SEQ) {
      int t_nxt = rev ? (T_SEQ - 2 - s) : (s + 1);
      __builtin_prefetch(Xg + (long)t_nxt * (B_SZ * GATES) + tid * 32, 0, 1);
    }

    // Hoist gate-bias loads ahead of the WMMA chain (no hbuf dependency):
    // their latency is hidden behind the 8x(ds_load + 4 WMMA) sequence.
    const float* xslice = Xg + (long)t_in * (B_SZ * GATES);
    float xg[4][8];
#pragma unroll
    for (int g = 0; g < 4; ++g) {
#pragma unroll
      for (int r = 0; r < 8; ++r) {
        int b = brow0 + r;
        xg[g][r] = xslice[(long)b * GATES + g * 256 + u];
      }
    }

    v8f acc[4];
#pragma unroll
    for (int g = 0; g < 4; ++g) { v8f z = {}; acc[g] = z; }

#pragma unroll
    for (int kt = 0; kt < 8; ++kt) {
      V16H a;
      const _Float16* ap = &hbuf[arowoff + kt * 32];
      a.q[0] = *(const u32x4*)(ap);
      a.q[1] = *(const u32x4*)(ap + 16);
#pragma unroll
      for (int g = 0; g < 4; ++g) {
        V16H b;
        const _Float16* bp = brow[g] + kt * 32;
        b.q[0] = *(const u32x4*)(bp);
        b.q[1] = *(const u32x4*)(bp + 16);
        acc[g] = __builtin_amdgcn_wmma_f32_16x16x32_f16(false, a.v, false, b.v,
                                                        (short)0, acc[g],
                                                        false, false);
      }
    }
    __syncthreads();   // all hbuf reads done before epilogue rewrites it

    float* yslice = y + (long)t_out * (B_SZ * UNITS);
#pragma unroll
    for (int r = 0; r < 8; ++r) {
      int b = brow0 + r;
      float gi = acc[0][r] + xg[0][r];
      float gf = acc[1][r] + xg[1][r];
      float gg = acc[2][r] + xg[2][r];
      float go = acc[3][r] + xg[3][r];
      float c  = sigmoidf_(gf) * creg[r] + sigmoidf_(gi) * tanhf(gg);
      float h  = sigmoidf_(go) * tanhf(c);
      creg[r] = c;
      hbuf[b * UNITS + u] = (_Float16)h;
      yslice[b * UNITS + u] = h;
    }
    __syncthreads();   // hbuf updated before next step's fragment loads
  }

  // final states
  float* hOut = stateOut + (long)dir * 2 * 8192;
  float* cOut = hOut + 8192;
#pragma unroll
  for (int r = 0; r < 8; ++r) {
    int b = brow0 + r;
    cOut[b * UNITS + u] = creg[r];
  }
  __syncthreads();
  for (int i = tid; i < B_SZ * UNITS; i += 1024)
    hOut[i] = (float)hbuf[i];
}

// ---------------------------------------------------------------------------
// Merge directions: y = yf + yb (both time-major [T][B][U]). Output is
// token-major [B][T][U]: f16 (next layer's GEMM input) or f32 (final output).
// ---------------------------------------------------------------------------
__global__ void combine_kernel(const float* __restrict__ yf,
                               const float* __restrict__ yb,
                               float* __restrict__ outF,
                               _Float16* __restrict__ outH, int n) {
  int i = blockIdx.x * blockDim.x + threadIdx.x;
  if (i >= n) return;
  int tok = i >> 8;              // /UNITS
  int u   = i & (UNITS - 1);
  int b   = tok >> 9;            // /T_SEQ
  int t   = tok & (T_SEQ - 1);
  long j  = ((long)t * B_SZ + b) * UNITS + u;
  float s = yf[j] + yb[j];
  if (outH) outH[i] = (_Float16)s;
  else      outF[i] = s;
}

// ---------------------------------------------------------------------------
extern "C" void kernel_launch(void* const* d_in, const int* in_sizes, int n_in,
                              void* d_out, int out_size, void* d_ws, size_t ws_size,
                              hipStream_t stream) {
  (void)in_sizes; (void)n_in; (void)out_size; (void)ws_size;

  const int*   x   = (const int*)d_in[0];
  const float* emb = (const float*)d_in[1];
  float* dout = (float*)d_out;

  char* ws = (char*)d_ws;
  size_t off = 0;
  auto alloc = [&](size_t bytes) -> void* {
    void* p = ws + off;
    off += (bytes + 255) & ~(size_t)255;
    return p;
  };

  _Float16* A16  = (_Float16*)alloc((size_t)NTOK * 320 * 2);  // layer input, f16
  float*    XgF  = (float*)   alloc((size_t)NTOK * GATES * 4);
  float*    XgB  = (float*)   alloc((size_t)NTOK * GATES * 4);
  _Float16* WiTF = (_Float16*)alloc((size_t)GATES * 320 * 2);
  _Float16* WiTB = (_Float16*)alloc((size_t)GATES * 320 * 2);
  _Float16* WhTF = (_Float16*)alloc((size_t)GATES * UNITS * 2);
  _Float16* WhTB = (_Float16*)alloc((size_t)GATES * UNITS * 2);
  float*    yF   = (float*)   alloc((size_t)NTOK * UNITS * 4);
  float*    yB   = (float*)   alloc((size_t)NTOK * UNITS * 4);
  float*    state= (float*)   alloc((size_t)4 * 8192 * 4);    // hf,cf,hb,cb

  // Embedding lookup -> f16 activations (K padded 300 -> 320)
  {
    int n = NTOK * 320;
    embed_cast_kernel<<<(n + 255) / 256, 256, 0, stream>>>(x, emb, A16, n);
  }

  for (int l = 0; l < 3; ++l) {
    const int K    = (l == 0) ? EMB_D : UNITS;
    const int Kpad = (l == 0) ? 320   : UNITS;
    const float* Wif = (const float*)d_in[2 + l * 6 + 0];
    const float* Whf = (const float*)d_in[2 + l * 6 + 1];
    const float* bfp = (const float*)d_in[2 + l * 6 + 2];
    const float* Wib = (const float*)d_in[2 + l * 6 + 3];
    const float* Whb = (const float*)d_in[2 + l * 6 + 4];
    const float* bbp = (const float*)d_in[2 + l * 6 + 5];

    // Weight repack: transpose + cast to f16, fragment-friendly layout
    {
      int nW = GATES * Kpad;
      wcast_kernel<<<(nW + 255) / 256, 256, 0, stream>>>(Wif, WiTF, K, GATES, Kpad);
      wcast_kernel<<<(nW + 255) / 256, 256, 0, stream>>>(Wib, WiTB, K, GATES, Kpad);
      int nWh = GATES * UNITS;
      wcast_kernel<<<(nWh + 255) / 256, 256, 0, stream>>>(Whf, WhTF, UNITS, GATES, UNITS);
      wcast_kernel<<<(nWh + 255) / 256, 256, 0, stream>>>(Whb, WhTB, UNITS, GATES, UNITS);
    }

    // Phase A: Xg = A @ Wi + b, both directions (blockIdx.y)
    {
      dim3 grid(1024, 2);   // 8192 wave-jobs x 8 tiles = 65536 16x16 tiles
      xproj_gemm_kernel<<<grid, 256, 0, stream>>>(A16, Kpad, WiTF, WiTB,
                                                  bfp, bbp, XgF, XgB, Kpad / 32);
    }

    // Phase B: recurrent scan, one persistent workgroup per direction
    {
      float* stOut = (l == 2) ? (dout + (size_t)NTOK * UNITS) : state;
      lstm_scan_kernel<<<2, 1024, 0, stream>>>(XgF, XgB, WhTF, WhTB,
                                               (l == 0) ? nullptr : state,
                                               yF, yB, stOut);
    }

    // Merge directions
    {
      int n = NTOK * UNITS;
      if (l < 2)
        combine_kernel<<<(n + 255) / 256, 256, 0, stream>>>(yF, yB, nullptr, A16, n);
      else
        combine_kernel<<<(n + 255) / 256, 256, 0, stream>>>(yF, yB, dout, nullptr, n);
    }
  }
}
